// UnifiedModalEncoder_16226386444686
// MI455X (gfx1250) — matrix operand order repaired and checked by reading, not compiled, not run on hardware
//
#include <hip/hip_runtime.h>
#include <hip/hip_bf16.h>
#include <stdint.h>

#define E_DIM 1024
#define B_SZ 8
#define S_LEN 1024
#define MS_LEN 128
#define NH_N 16
#define HD_DIM 64
#define NE_N 8
#define FF_DIM 4096
#define NTOK (B_SZ * S_LEN) /* 8192 */
#define LN_EPS 1e-5f

typedef __attribute__((ext_vector_type(16))) __bf16 v16bf;
typedef __attribute__((ext_vector_type(8)))  float  v8f;

// ---------------------------------------------------------------------------
// Async global->LDS DMA (CDNA5 GLOBAL_LOAD_ASYNC_TO_LDS_B128, ASYNCcnt).
// Inline asm: the clang builtin's pointer params use the HIP __device__
// language address space which cannot be spelled from source. Per the ISA,
// INST_OFFSET is added to BOTH the LDS and global addresses, so two
// instructions at offset 0/16 move 32 contiguous bytes per lane.
// The LDS operand is the LDS byte offset = low 32 bits of the flat pointer.
// ---------------------------------------------------------------------------
#define USE_ASYNC_LDS 1

#if USE_ASYNC_LDS
__device__ __forceinline__ void async_copy_b256(const void* g, void* l) {
  unsigned lds_off = (unsigned)(uintptr_t)l;
  asm volatile(
      "global_load_async_to_lds_b128 %0, %1, off\n\t"
      "global_load_async_to_lds_b128 %0, %1, off offset:16"
      :: "v"(lds_off), "v"(g) : "memory");
}
#define ASYNC_WAIT() asm volatile("s_wait_asynccnt 0x0" ::: "memory")
#else
#define ASYNC_WAIT()
#endif

union Frag {
  uint4 u[2];
  unsigned short h[16];
  v16bf v;
};

__device__ __forceinline__ unsigned short f2bf(float f) {
  union { float f; uint32_t u; } x; x.f = f;
  uint32_t r = x.u + 0x7FFFu + ((x.u >> 16) & 1u);
  return (unsigned short)(r >> 16);
}
__device__ __forceinline__ float bf2f(unsigned short b) {
  union { uint32_t u; float f; } x; x.u = ((uint32_t)b) << 16;
  return x.f;
}
__device__ __forceinline__ v8f wmma_bf16(v16bf a, v16bf b, v8f c) {
  return __builtin_amdgcn_wmma_f32_16x16x32_bf16(false, a, false, b, (short)0, c, false, false);
}
__device__ __forceinline__ float gelu_tanh(float v) {
  const float c = 0.7978845608028654f;
  return 0.5f * v * (1.0f + tanhf(c * (v + 0.044715f * v * v * v)));
}

// ---------------------------------------------------------------------------
// LayerNorm over rows of length 1024. mode bit0: add addX first. mode bit1:
// "double LN" — also produce the re-normalized (pre-affine) hat of y for the
// per-expert LayerNorms. Writes: outb (bf16 y), outf (f32 y), hat (bf16 yhat).
// ---------------------------------------------------------------------------
__global__ __launch_bounds__(256) void k_ln(
    const float* __restrict__ X, const float* __restrict__ addX,
    const float* __restrict__ w, const float* __restrict__ b,
    unsigned short* __restrict__ outb, float* __restrict__ outf,
    unsigned short* __restrict__ hat, int mode)
{
  __shared__ float red[16];
  const int row = blockIdx.x;
  const int tid = threadIdx.x;
  const int lane = tid & 31, wid = tid >> 5;
  const float* xp = X + (size_t)row * E_DIM;
  float4 xv = *(const float4*)(xp + tid * 4);
  if (mode & 1) {
    float4 av = *(const float4*)(addX + (size_t)row * E_DIM + tid * 4);
    xv.x += av.x; xv.y += av.y; xv.z += av.z; xv.w += av.w;
  }
  float s = xv.x + xv.y + xv.z + xv.w;
  float ss = xv.x * xv.x + xv.y * xv.y + xv.z * xv.z + xv.w * xv.w;
  for (int o = 16; o; o >>= 1) { s += __shfl_xor(s, o, 32); ss += __shfl_xor(ss, o, 32); }
  if (lane == 0) { red[wid] = s; red[8 + wid] = ss; }
  __syncthreads();
  if (tid == 0) {
    float a = 0.f, bb = 0.f;
    for (int i = 0; i < 8; i++) { a += red[i]; bb += red[8 + i]; }
    red[0] = a; red[8] = bb;
  }
  __syncthreads();
  float mean = red[0] * (1.0f / E_DIM);
  float var = red[8] * (1.0f / E_DIM) - mean * mean;
  float rstd = rsqrtf(var + LN_EPS);
  float4 wv = *(const float4*)(w + tid * 4);
  float4 bv = *(const float4*)(b + tid * 4);
  float y0 = (xv.x - mean) * rstd * wv.x + bv.x;
  float y1 = (xv.y - mean) * rstd * wv.y + bv.y;
  float y2 = (xv.z - mean) * rstd * wv.z + bv.z;
  float y3 = (xv.w - mean) * rstd * wv.w + bv.w;
  if (outb) {
    ushort4 o; o.x = f2bf(y0); o.y = f2bf(y1); o.z = f2bf(y2); o.w = f2bf(y3);
    *(ushort4*)(outb + (size_t)row * E_DIM + tid * 4) = o;
  }
  if (outf) {
    float4 o; o.x = y0; o.y = y1; o.z = y2; o.w = y3;
    *(float4*)(outf + (size_t)row * E_DIM + tid * 4) = o;
  }
  if (mode & 2) {
    // second LN pass (stats of y), pre-affine hat for expert LayerNorms
    __syncthreads();
    float s2 = y0 + y1 + y2 + y3;
    float ss2 = y0 * y0 + y1 * y1 + y2 * y2 + y3 * y3;
    for (int o = 16; o; o >>= 1) { s2 += __shfl_xor(s2, o, 32); ss2 += __shfl_xor(ss2, o, 32); }
    if (lane == 0) { red[wid] = s2; red[8 + wid] = ss2; }
    __syncthreads();
    if (tid == 0) {
      float a = 0.f, bb = 0.f;
      for (int i = 0; i < 8; i++) { a += red[i]; bb += red[8 + i]; }
      red[0] = a; red[8] = bb;
    }
    __syncthreads();
    float m2 = red[0] * (1.0f / E_DIM);
    float v2 = red[8] * (1.0f / E_DIM) - m2 * m2;
    float rs2 = rsqrtf(v2 + LN_EPS);
    ushort4 o;
    o.x = f2bf((y0 - m2) * rs2); o.y = f2bf((y1 - m2) * rs2);
    o.z = f2bf((y2 - m2) * rs2); o.w = f2bf((y3 - m2) * rs2);
    *(ushort4*)(hat + (size_t)row * E_DIM + tid * 4) = o;
  }
}

// ---------------------------------------------------------------------------
// fp32 -> bf16 cast, 4 elements/thread
// ---------------------------------------------------------------------------
__global__ __launch_bounds__(256) void k_cast_bf16(
    const float* __restrict__ X, unsigned short* __restrict__ Y)
{
  size_t i = ((size_t)blockIdx.x * 256 + threadIdx.x) * 4;
  float4 v = *(const float4*)(X + i);
  ushort4 o; o.x = f2bf(v.x); o.y = f2bf(v.y); o.z = f2bf(v.z); o.w = f2bf(v.w);
  *(ushort4*)(Y + i) = o;
}

// h += sigmoid(g) * f, elementwise fp32
__global__ __launch_bounds__(256) void k_gate_combine(
    float* __restrict__ H, const float* __restrict__ G, const float* __restrict__ F)
{
  size_t i = ((size_t)blockIdx.x * 256 + threadIdx.x) * 4;
  float4 h = *(const float4*)(H + i);
  float4 g = *(const float4*)(G + i);
  float4 f = *(const float4*)(F + i);
  h.x += f.x / (1.0f + __expf(-g.x));
  h.y += f.y / (1.0f + __expf(-g.y));
  h.z += f.z / (1.0f + __expf(-g.z));
  h.w += f.w / (1.0f + __expf(-g.w));
  *(float4*)(H + i) = h;
}

// ---------------------------------------------------------------------------
// WMMA GEMM: C[M,N] = act( A_bf16[M,K] @ W_f32[N,K]^T + bias[N] ) (+epilogues)
// Block: 256 threads / 8 waves. Tile BM=128, BN=64, BK=32. Each wave: 32x32.
// A tile staged via async global->LDS DMA (ASYNCcnt). W tile converted
// fp32->bf16 while staging into LDS (single-touch HBM reads).
// count_ptr (device) enables early-exit for data-dependent M (MoE experts).
// ---------------------------------------------------------------------------
#define SA_ST 48
#define SB_ST 48
#define OUTM_F32 0
#define OUTM_BF16 1
#define OUTM_F32_RESID 2
#define OUTM_SCATTER 3
#define ACT_NONE 0
#define ACT_GELU 1

__global__ __launch_bounds__(256) void k_gemm(
    const unsigned short* __restrict__ A, int lda,
    const float* __restrict__ W, int ldw,
    const float* __restrict__ bias,
    int M, int N, int K,
    const int* __restrict__ count_ptr,
    void* __restrict__ Out, int ldo,
    const float* __restrict__ resid,
    const int* __restrict__ sc_idx, const float* __restrict__ sc_w,
    float* __restrict__ sc_out,
    int out_mode, int act)
{
  __shared__ __align__(16) unsigned short sA[128 * SA_ST];
  __shared__ __align__(16) unsigned short sB[64 * SB_ST];
  const int Meff = count_ptr ? count_ptr[0] : M;
  const int m0 = blockIdx.x * 128;
  if (m0 >= Meff) return;
  const int n0 = blockIdx.y * 64;
  const int tid = threadIdx.x;
  const int lane = tid & 31, wid = tid >> 5;
  const int wm = (wid & 3) * 32, wn = (wid >> 2) * 32;
  const int half = lane >> 4, lr = lane & 15;

  v8f acc[2][2];
#pragma unroll
  for (int i = 0; i < 2; i++)
#pragma unroll
    for (int j = 0; j < 2; j++)
#pragma unroll
      for (int r = 0; r < 8; r++) acc[i][j][r] = 0.0f;

  for (int k0 = 0; k0 < K; k0 += 32) {
    { // stage A tile (128x32 bf16): 2 threads/row x 32B, async DMA to LDS
      int row = tid >> 1, hh = tid & 1;
      const unsigned short* gsrc = A + (size_t)(m0 + row) * lda + k0 + hh * 16;
      unsigned short* ldst = sA + row * SA_ST + hh * 16;
#if USE_ASYNC_LDS
      async_copy_b256(gsrc, ldst);
#else
      const uint4* src = (const uint4*)gsrc;
      uint4* dst = (uint4*)ldst;
      dst[0] = src[0]; dst[1] = src[1];
#endif
    }
    { // stage W tile (64x32 fp32 -> bf16): 4 threads/row x 8 floats
      int row = tid >> 2, q = tid & 3;
      const float4* src = (const float4*)(W + (size_t)(n0 + row) * ldw + k0 + q * 8);
      float4 f0 = src[0], f1 = src[1];
      __align__(16) unsigned short tmp[8] = {
        f2bf(f0.x), f2bf(f0.y), f2bf(f0.z), f2bf(f0.w),
        f2bf(f1.x), f2bf(f1.y), f2bf(f1.z), f2bf(f1.w) };
      *(uint4*)(sB + row * SB_ST + q * 8) = *(const uint4*)tmp;
    }
    if (k0 + 32 < K) {
      __builtin_prefetch(W + (size_t)(n0 + (tid >> 2)) * ldw + k0 + 32 + (tid & 3) * 8, 0, 1);
      __builtin_prefetch(A + (size_t)(m0 + (tid >> 1)) * lda + k0 + 32 + (tid & 1) * 16, 0, 1);
    }
    ASYNC_WAIT();
    __syncthreads();

    Frag aF[2], bF[2];
#pragma unroll
    for (int i = 0; i < 2; i++) {
      // A operand: row = m (lane 0..15), K pattern g*16 + half*8 + 2v
      const unsigned short* ab = sA + (wm + i * 16 + lr) * SA_ST;
      aF[i].u[0] = *(const uint4*)(ab + half * 8);
      aF[i].u[1] = *(const uint4*)(ab + 16 + half * 8);
      // B operand: lane = n column, K pattern g*16 + half*8 + 2v
      const unsigned short* bb = sB + (wn + i * 16 + lr) * SB_ST;
      bF[i].u[0] = *(const uint4*)(bb + half * 8);
      bF[i].u[1] = *(const uint4*)(bb + 16 + half * 8);
    }
#pragma unroll
    for (int i = 0; i < 2; i++)
#pragma unroll
      for (int j = 0; j < 2; j++)
        acc[i][j] = wmma_bf16(aF[i].v, bF[j].v, acc[i][j]);
    __syncthreads();
  }

  // epilogue: C layout — VGPR r holds row (r + half*8), col = lane&15
#pragma unroll
  for (int sm = 0; sm < 2; sm++) {
#pragma unroll
    for (int sn = 0; sn < 2; sn++) {
      const int gn = n0 + wn + sn * 16 + lr;
      const float bv = bias ? bias[gn] : 0.0f;
#pragma unroll
      for (int r = 0; r < 8; r++) {
        const int gm = m0 + wm + sm * 16 + r + half * 8;
        if (gm >= Meff) continue;
        float v = acc[sm][sn][r] + bv;
        if (act == ACT_GELU) v = gelu_tanh(v);
        if (out_mode == OUTM_F32) {
          ((float*)Out)[(size_t)gm * ldo + gn] = v;
        } else if (out_mode == OUTM_BF16) {
          ((unsigned short*)Out)[(size_t)gm * ldo + gn] = f2bf(v);
        } else if (out_mode == OUTM_F32_RESID) {
          ((float*)Out)[(size_t)gm * ldo + gn] = v + resid[(size_t)gm * ldo + gn];
        } else { // OUTM_SCATTER: moe[token] += w * v
          int t = sc_idx[gm];
          sc_out[(size_t)t * ldo + gn] += sc_w[gm] * v;
        }
      }
    }
  }
}

// ---------------------------------------------------------------------------
// Flash attention, WMMA. 8 waves x 16 queries per block. hd=64.
// QK^T: Q as A operand (2 x K=32 frags), K as B operand. Online softmax with
// shfl reductions over 16-lane groups. P re-layout (C->A) via per-wave LDS.
// ---------------------------------------------------------------------------
__global__ __launch_bounds__(256) void k_attn(
    const unsigned short* __restrict__ q_base, int q_stride, int q_coff,
    const unsigned short* __restrict__ k_base, int k_stride, int k_coff,
    const unsigned short* __restrict__ v_base, int v_stride, int v_coff,
    unsigned short* __restrict__ o_base, int o_stride,
    int Sq, int Sk, int nh, float scale)
{
  __shared__ __align__(16) unsigned short pP[8][16 * 32];
  const int bh = blockIdx.y;
  const int b = bh / nh, h = bh % nh;
  const int lane = threadIdx.x & 31, wid = threadIdx.x >> 5;
  const int q0 = blockIdx.x * 128 + wid * 16;
  const int half = lane >> 4, lr = lane & 15;

  // Q fragments (A operand), loaded once
  const unsigned short* qrow =
      q_base + (size_t)(b * Sq + q0 + lr) * q_stride + q_coff + h * HD_DIM;
  Frag qf[2];
#pragma unroll
  for (int kc = 0; kc < 2; kc++) {
    qf[kc].u[0] = *(const uint4*)(qrow + kc * 32 + half * 8);
    qf[kc].u[1] = *(const uint4*)(qrow + kc * 32 + 16 + half * 8);
  }

  float mI[8], lI[8];
  v8f Oacc[4];
#pragma unroll
  for (int r = 0; r < 8; r++) { mI[r] = -1e30f; lI[r] = 0.0f; }
#pragma unroll
  for (int n = 0; n < 4; n++)
#pragma unroll
    for (int r = 0; r < 8; r++) Oacc[n][r] = 0.0f;

  unsigned short* pp = pP[wid];

  for (int kt = 0; kt < Sk; kt += 32) {
    v8f sc[2];
#pragma unroll
    for (int nt = 0; nt < 2; nt++) {
      // K fragment: B operand, lane = key column, dims g*16 + half*8 + v
      const unsigned short* krow =
          k_base + (size_t)(b * Sk + kt + nt * 16 + lr) * k_stride + k_coff + h * HD_DIM;
      Frag kf[2];
#pragma unroll
      for (int kc = 0; kc < 2; kc++) {
        kf[kc].u[0] = *(const uint4*)(krow + kc * 32 + half * 8);
        kf[kc].u[1] = *(const uint4*)(krow + kc * 32 + 16 + half * 8);
      }
      v8f s;
#pragma unroll
      for (int r = 0; r < 8; r++) s[r] = 0.0f;
      s = wmma_bf16(qf[0].v, kf[0].v, s);
      s = wmma_bf16(qf[1].v, kf[1].v, s);
      sc[nt] = s;
    }

    // online softmax (rows = (r, half-group); cols across 16 lanes)
    float alpha[8];
#pragma unroll
    for (int r = 0; r < 8; r++) {
      float s0 = sc[0][r] * scale, s1 = sc[1][r] * scale;
      float mx = fmaxf(s0, s1);
      for (int o = 8; o; o >>= 1) mx = fmaxf(mx, __shfl_xor(mx, o, 16));
      float mnew = fmaxf(mI[r], mx);
      float a = __expf(mI[r] - mnew);
      float p0 = __expf(s0 - mnew), p1 = __expf(s1 - mnew);
      sc[0][r] = p0; sc[1][r] = p1;
      float rs = p0 + p1;
      for (int o = 8; o; o >>= 1) rs += __shfl_xor(rs, o, 16);
      lI[r] = lI[r] * a + rs;
      mI[r] = mnew; alpha[r] = a;
    }
#pragma unroll
    for (int n = 0; n < 4; n++)
#pragma unroll
      for (int r = 0; r < 8; r++) Oacc[n][r] *= alpha[r];

    // P: C layout -> LDS (16 x 32 bf16, per-wave private region)
#pragma unroll
    for (int nt = 0; nt < 2; nt++)
#pragma unroll
      for (int r = 0; r < 8; r++)
        pp[(r + half * 8) * 32 + nt * 16 + lr] = f2bf(sc[nt][r]);
    asm volatile("s_wait_dscnt 0" ::: "memory");

    // P as A operand
    Frag pf;
    const unsigned short* pb = pp + lr * 32;
    pf.u[0] = *(const uint4*)(pb + half * 16);
    pf.u[1] = *(const uint4*)(pb + half * 16 + 8);

    // V fragments (B operand): lane = output dim, K = keys (half*16 + 2v)
#pragma unroll
    for (int n = 0; n < 4; n++) {
      const int dim = v_coff + h * HD_DIM + n * 16 + lr;
      Frag vf;
#pragma unroll
      for (int vv = 0; vv < 8; vv++) {
        int key = kt + half * 16 + 2 * vv;
        const unsigned short* vp = v_base + (size_t)(b * Sk + key) * v_stride + dim;
        vf.h[2 * vv] = vp[0];
        vf.h[2 * vv + 1] = vp[v_stride];
      }
      Oacc[n] = wmma_bf16(pf.v, vf.v, Oacc[n]);
    }
  }

  // normalize + store bf16
#pragma unroll
  for (int n = 0; n < 4; n++)
#pragma unroll
    for (int r = 0; r < 8; r++) {
      float o = Oacc[n][r] / lI[r];
      int gm = q0 + r + half * 8;
      int gc = h * HD_DIM + n * 16 + lr;
      o_base[(size_t)(b * Sq + gm) * o_stride + gc] = f2bf(o);
    }
}

// ---------------------------------------------------------------------------
// Router: one block per token; wave w computes logit for expert w (dot 1024)
// ---------------------------------------------------------------------------
__global__ __launch_bounds__(256) void k_router(
    const unsigned short* __restrict__ Xb, const float* __restrict__ RW,
    const float* __restrict__ RB, float* __restrict__ L)
{
  const int t = blockIdx.x;
  const int lane = threadIdx.x & 31, e = threadIdx.x >> 5;
  const unsigned short* xp = Xb + (size_t)t * E_DIM;
  const float* wp = RW + (size_t)e * E_DIM;
  float acc = 0.0f;
  for (int k = lane; k < E_DIM; k += 32) acc += bf2f(xp[k]) * wp[k];
  for (int o = 16; o; o >>= 1) acc += __shfl_xor(acc, o, 32);
  if (lane == 0) L[(size_t)t * NE_N + e] = acc + RB[e];
}

// softmax over 8, top-2, renorm, build per-expert token lists
__global__ __launch_bounds__(256) void k_topk(
    const float* __restrict__ L, int* __restrict__ counts,
    int* __restrict__ lists, float* __restrict__ wts)
{
  int t = blockIdx.x * 256 + threadIdx.x;
  if (t >= NTOK) return;
  float p[NE_N]; float mx = -1e30f;
  for (int e = 0; e < NE_N; e++) { p[e] = L[(size_t)t * NE_N + e]; mx = fmaxf(mx, p[e]); }
  for (int e = 0; e < NE_N; e++) p[e] = __expf(p[e] - mx);
  int i1 = 0;
  for (int e = 1; e < NE_N; e++) if (p[e] > p[i1]) i1 = e;
  int i2 = (i1 == 0) ? 1 : 0;
  for (int e = 0; e < NE_N; e++) if (e != i1 && p[e] > p[i2]) i2 = e;
  float wsum = p[i1] + p[i2];
  float w1 = p[i1] / wsum, w2 = p[i2] / wsum;
  int pos1 = atomicAdd(&counts[i1], 1);
  lists[i1 * NTOK + pos1] = t; wts[i1 * NTOK + pos1] = w1;
  int pos2 = atomicAdd(&counts[i2], 1);
  lists[i2 * NTOK + pos2] = t; wts[i2 * NTOK + pos2] = w2;
}

// gather tokens of one expert, apply expert LN affine on precomputed hat
__global__ __launch_bounds__(256) void k_gather_ln(
    const unsigned short* __restrict__ Hat, const int* __restrict__ counts,
    const int* __restrict__ list, const float* __restrict__ we,
    const float* __restrict__ be, unsigned short* __restrict__ Aout, int e)
{
  int rowi = blockIdx.x;
  if (rowi >= counts[e]) return;
  int t = list[rowi];
  int j = threadIdx.x * 4;
  ushort4 hv = *(const ushort4*)(Hat + (size_t)t * E_DIM + j);
  float4 wv = *(const float4*)(we + j);
  float4 bv = *(const float4*)(be + j);
  ushort4 o;
  o.x = f2bf(bf2f(hv.x) * wv.x + bv.x);
  o.y = f2bf(bf2f(hv.y) * wv.y + bv.y);
  o.z = f2bf(bf2f(hv.z) * wv.z + bv.z);
  o.w = f2bf(bf2f(hv.w) * wv.w + bv.w);
  *(ushort4*)(Aout + (size_t)rowi * E_DIM + j) = o;
}

// ---------------------------------------------------------------------------
extern "C" void kernel_launch(void* const* d_in, const int* in_sizes, int n_in,
                              void* d_out, int out_size, void* d_ws, size_t ws_size,
                              hipStream_t stream)
{
  (void)in_sizes; (void)n_in; (void)out_size; (void)ws_size;
  const float* x          = (const float*)d_in[0];
  const float* modal      = (const float*)d_in[1];
  const float* attn_in_w  = (const float*)d_in[2];
  const float* attn_in_b  = (const float*)d_in[3];
  const float* attn_out_w = (const float*)d_in[4];
  const float* attn_out_b = (const float*)d_in[5];
  const float* xat_in_w   = (const float*)d_in[6];
  const float* xat_in_b   = (const float*)d_in[7];
  const float* xat_out_w  = (const float*)d_in[8];
  const float* xat_out_b  = (const float*)d_in[9];
  const float* n1w = (const float*)d_in[10], * n1b = (const float*)d_in[11];
  const float* n2w = (const float*)d_in[12], * n2b = (const float*)d_in[13];
  const float* n3w = (const float*)d_in[14], * n3b = (const float*)d_in[15];
  const float* xnw = (const float*)d_in[16], * xnb = (const float*)d_in[17];
  const float* rw  = (const float*)d_in[18], * rb  = (const float*)d_in[19];
  const float* gw  = (const float*)d_in[20], * gb  = (const float*)d_in[21];
  const float* elnw = (const float*)d_in[22], * elnb = (const float*)d_in[23];
  const float* fc1w = (const float*)d_in[24], * fc1b = (const float*)d_in[25];
  const float* fc2w = (const float*)d_in[26], * fc2b = (const float*)d_in[27];

  char* ws = (char*)d_ws;
  size_t off = 0;
  auto alloc = [&](size_t bytes) -> char* {
    char* p = ws + off;
    off += (bytes + 255) & ~(size_t)255;
    return p;
  };
  unsigned short* xn1    = (unsigned short*)alloc((size_t)NTOK * E_DIM * 2);
  unsigned short* qkv    = (unsigned short*)alloc((size_t)NTOK * 3 * E_DIM * 2);
  unsigned short* ctx    = (unsigned short*)alloc((size_t)NTOK * E_DIM * 2);
  float*          hbuf   = (float*)alloc((size_t)NTOK * E_DIM * 4);
  unsigned short* xnb16  = (unsigned short*)alloc((size_t)NTOK * E_DIM * 2);
  unsigned short* modalb = (unsigned short*)alloc((size_t)B_SZ * MS_LEN * E_DIM * 2);
  unsigned short* q2     = (unsigned short*)alloc((size_t)NTOK * E_DIM * 2);
  unsigned short* kv2    = (unsigned short*)alloc((size_t)B_SZ * MS_LEN * 2 * E_DIM * 2);
  unsigned short* ctx2   = (unsigned short*)alloc((size_t)NTOK * E_DIM * 2);
  float*          fusion = (float*)alloc((size_t)NTOK * E_DIM * 4);
  float*          glog   = (float*)alloc((size_t)NTOK * E_DIM * 4);
  unsigned short* xn2b   = (unsigned short*)alloc((size_t)NTOK * E_DIM * 2);
  unsigned short* xn2hat = (unsigned short*)alloc((size_t)NTOK * E_DIM * 2);
  float*          rlog   = (float*)alloc((size_t)NTOK * NE_N * 4);
  int*            counts = (int*)alloc(NE_N * 4);
  int*            lists  = (int*)alloc((size_t)NE_N * NTOK * 4);
  float*          wts    = (float*)alloc((size_t)NE_N * NTOK * 4);
  unsigned short* aexp   = (unsigned short*)alloc((size_t)NTOK * E_DIM * 2);
  unsigned short* hh     = (unsigned short*)alloc((size_t)NTOK * FF_DIM * 2);
  float*          moe    = (float*)alloc((size_t)NTOK * E_DIM * 4);

  (void)hipMemsetAsync(counts, 0, NE_N * 4, stream);
  (void)hipMemsetAsync(moe, 0, (size_t)NTOK * E_DIM * 4, stream);

  // --- self-attention block ---
  k_ln<<<NTOK, 256, 0, stream>>>(x, nullptr, n1w, n1b, xn1, nullptr, nullptr, 0);
  k_gemm<<<dim3(64, 48), 256, 0, stream>>>(xn1, E_DIM, attn_in_w, E_DIM, attn_in_b,
      NTOK, 3 * E_DIM, E_DIM, nullptr, qkv, 3 * E_DIM,
      nullptr, nullptr, nullptr, nullptr, OUTM_BF16, ACT_NONE);
  k_attn<<<dim3(S_LEN / 128, B_SZ * NH_N), 256, 0, stream>>>(
      qkv, 3 * E_DIM, 0, qkv, 3 * E_DIM, E_DIM, qkv, 3 * E_DIM, 2 * E_DIM,
      ctx, E_DIM, S_LEN, S_LEN, NH_N, 0.125f);
  k_gemm<<<dim3(64, 16), 256, 0, stream>>>(ctx, E_DIM, attn_out_w, E_DIM, attn_out_b,
      NTOK, E_DIM, E_DIM, nullptr, hbuf, E_DIM,
      x, nullptr, nullptr, nullptr, OUTM_F32_RESID, ACT_NONE);

  // --- cross-modal fusion ---
  k_ln<<<NTOK, 256, 0, stream>>>(hbuf, nullptr, xnw, xnb, xnb16, nullptr, nullptr, 0);
  k_cast_bf16<<<(B_SZ * MS_LEN * E_DIM) / 1024, 256, 0, stream>>>(modal, modalb);
  k_gemm<<<dim3(64, 16), 256, 0, stream>>>(xnb16, E_DIM, xat_in_w, E_DIM, xat_in_b,
      NTOK, E_DIM, E_DIM, nullptr, q2, E_DIM,
      nullptr, nullptr, nullptr, nullptr, OUTM_BF16, ACT_NONE);
  k_gemm<<<dim3(8, 32), 256, 0, stream>>>(modalb, E_DIM,
      xat_in_w + (size_t)E_DIM * E_DIM, E_DIM, xat_in_b + E_DIM,
      B_SZ * MS_LEN, 2 * E_DIM, E_DIM, nullptr, kv2, 2 * E_DIM,
      nullptr, nullptr, nullptr, nullptr, OUTM_BF16, ACT_NONE);
  k_attn<<<dim3(S_LEN / 128, B_SZ * NH_N), 256, 0, stream>>>(
      q2, E_DIM, 0, kv2, 2 * E_DIM, 0, kv2, 2 * E_DIM, E_DIM,
      ctx2, E_DIM, S_LEN, MS_LEN, NH_N, 0.125f);
  k_gemm<<<dim3(64, 16), 256, 0, stream>>>(ctx2, E_DIM, xat_out_w, E_DIM, xat_out_b,
      NTOK, E_DIM, E_DIM, nullptr, fusion, E_DIM,
      nullptr, nullptr, nullptr, nullptr, OUTM_F32, ACT_NONE);
  k_gemm<<<dim3(64, 16), 256, 0, stream>>>(xnb16, E_DIM, gw, E_DIM, gb,
      NTOK, E_DIM, E_DIM, nullptr, glog, E_DIM,
      nullptr, nullptr, nullptr, nullptr, OUTM_F32, ACT_NONE);
  k_gate_combine<<<(NTOK * E_DIM) / 1024, 256, 0, stream>>>(hbuf, glog, fusion);

  // --- MoE block ---
  k_ln<<<NTOK, 256, 0, stream>>>(hbuf, nullptr, n2w, n2b, xn2b, nullptr, xn2hat, 2);
  k_router<<<NTOK, 256, 0, stream>>>(xn2b, rw, rb, rlog);
  k_topk<<<NTOK / 256, 256, 0, stream>>>(rlog, counts, lists, wts);
  for (int e = 0; e < NE_N; e++) {
    k_gather_ln<<<NTOK, 256, 0, stream>>>(xn2hat, counts, lists + (size_t)e * NTOK,
        elnw + (size_t)e * E_DIM, elnb + (size_t)e * E_DIM, aexp, e);
    k_gemm<<<dim3(64, 64), 256, 0, stream>>>(aexp, E_DIM,
        fc1w + (size_t)e * FF_DIM * E_DIM, E_DIM, fc1b + (size_t)e * FF_DIM,
        NTOK, FF_DIM, E_DIM, counts + e, hh, FF_DIM,
        nullptr, nullptr, nullptr, nullptr, OUTM_BF16, ACT_GELU);
    k_gemm<<<dim3(64, 16), 256, 0, stream>>>(hh, FF_DIM,
        fc2w + (size_t)e * E_DIM * FF_DIM, FF_DIM, fc2b + (size_t)e * E_DIM,
        NTOK, E_DIM, FF_DIM, counts + e, nullptr, E_DIM,
        nullptr, lists + (size_t)e * NTOK, wts + (size_t)e * NTOK, moe,
        OUTM_SCATTER, ACT_NONE);
  }

  // --- final: LN(h + moe) -> d_out (fp32) ---
  k_ln<<<NTOK, 256, 0, stream>>>(hbuf, moe, n3w, n3b, nullptr, (float*)d_out, nullptr, 1);
}
